// EdgeSelfAttnBlock_26594437497098
// MI455X (gfx1250) — matrix operand order repaired and checked by reading, not compiled, not run on hardware
//
#include <hip/hip_runtime.h>
#include <hip/hip_bf16.h>
#include <math.h>

#define D_MODEL 128
#define N_HEADS 8
#define DH 16
#define EDGE_F 16
#define EB_HID 32
#define FFN_HID 512
#define N_NODES 50000
#define N_EDGES 800000

typedef __attribute__((ext_vector_type(16))) __bf16        v16bf;
typedef __attribute__((ext_vector_type(8)))  float         v8f;
typedef __attribute__((ext_vector_type(8)))  unsigned short us8;
typedef __attribute__((ext_vector_type(4)))  unsigned short us4;

__device__ __forceinline__ unsigned short f2bf(float f) {
    unsigned int u = __float_as_uint(f);
    u += 0x7fffu + ((u >> 16) & 1u);           // round-to-nearest-even
    return (unsigned short)(u >> 16);
}

__device__ __forceinline__ float wave_sum32(float v) {
    #pragma unroll
    for (int o = 16; o > 0; o >>= 1) v += __shfl_xor(v, o, 32);
    return v;
}

// ---------------------------------------------------------------------------
// Transpose + f32 -> bf16 weight prep:  W (R x C, row-major) -> WT (C x R)
// ---------------------------------------------------------------------------
__global__ void transpose_to_bf16_kernel(const float* __restrict__ W,
                                         unsigned short* __restrict__ WT,
                                         int R, int C) {
    int i = blockIdx.x * blockDim.x + threadIdx.x;
    if (i >= R * C) return;
    int r = i / C, c = i % C;
    WT[(size_t)c * R + r] = f2bf(W[(size_t)r * C + c]);
}

// ---------------------------------------------------------------------------
// LayerNorm (one wave32 per node, 4 elems/lane) -> bf16 output
// ---------------------------------------------------------------------------
__global__ void ln_bf16_kernel(const float* __restrict__ x,
                               const float* __restrict__ g,
                               const float* __restrict__ b,
                               unsigned short* __restrict__ y,
                               int nNodes) {
    int wid  = threadIdx.x >> 5;
    int lane = threadIdx.x & 31;
    int node = blockIdx.x * (blockDim.x >> 5) + wid;
    if (node >= nNodes) return;
    const float4* xp = (const float4*)(x + (size_t)node * D_MODEL);
    float4 v = xp[lane];
    float mean = wave_sum32(v.x + v.y + v.z + v.w) * (1.0f / D_MODEL);
    float dx0 = v.x - mean, dx1 = v.y - mean, dx2 = v.z - mean, dx3 = v.w - mean;
    float var = wave_sum32(dx0*dx0 + dx1*dx1 + dx2*dx2 + dx3*dx3) * (1.0f / D_MODEL);
    float rstd = rsqrtf(var + 1e-5f);
    float4 gg = ((const float4*)g)[lane];
    float4 bb = ((const float4*)b)[lane];
    us4 o;
    o.x = f2bf(dx0 * rstd * gg.x + bb.x);
    o.y = f2bf(dx1 * rstd * gg.y + bb.y);
    o.z = f2bf(dx2 * rstd * gg.z + bb.z);
    o.w = f2bf(dx3 * rstd * gg.w + bb.w);
    ((us4*)y)[(size_t)node * 32 + lane] = o;
}

// ---------------------------------------------------------------------------
// WMMA bf16 GEMM:  C[M x N] = A[M x K] * BT[N x K]^T  (+bias)(+act)(+resid)
// One wave per 16x64 tile (4 accumulators): A fragment loaded once per
// k-step and reused by 4 WMMAs -> 2.5 b128 loads per v_wmma instead of 4.
// Fragment layouts per CDNA5 ISA 7.12.2.
// ---------------------------------------------------------------------------
union FragU { us8 u[2]; v16bf v; };

template<int ACT, int RES>
__global__ void gemm_bf16_wmma_kernel(const unsigned short* __restrict__ A,
                                      const unsigned short* __restrict__ BT,
                                      const float* __restrict__ bias,
                                      const float* __restrict__ resid,
                                      float* __restrict__ Cf,
                                      unsigned short* __restrict__ Cb,
                                      int M, int N, int K) {
    int wave = threadIdx.x >> 5;
    int lane = threadIdx.x & 31;
    int tilesN = N >> 6;                         // 64-wide column tiles
    int tile = blockIdx.x * (blockDim.x >> 5) + wave;
    if (tile >= (M >> 4) * tilesN) return;
    int tm = tile / tilesN, tn = tile % tilesN;
    int l16 = lane & 15, hi = lane >> 4;

    size_t rowA = (size_t)(tm * 16 + l16) * K;   // A: lane holds row M=l16
    int aoff = hi * 8;                           // A: K lo/hi 8 split
    int boff = hi * 8;                           // B: K lo/hi 8 split
    size_t rowB[4];
    #pragma unroll
    for (int j = 0; j < 4; ++j)                  // B: lane holds col N=l16
        rowB[j] = (size_t)(tn * 64 + j * 16 + l16) * K;

    v8f acc[4] = {};
    for (int k0 = 0; k0 < K; k0 += 32) {
        FragU fa;
        fa.u[0] = *(const us8*)(A + rowA + k0 + aoff);        // K = aoff..aoff+7
        fa.u[1] = *(const us8*)(A + rowA + k0 + 16 + aoff);   // K = 16+aoff..
        #pragma unroll
        for (int j = 0; j < 4; ++j) {
            FragU fb;
            fb.u[0] = *(const us8*)(BT + rowB[j] + k0 + boff); // K = boff..boff+7
            fb.u[1] = *(const us8*)(BT + rowB[j] + k0 + 16 + boff);
            acc[j] = __builtin_amdgcn_wmma_f32_16x16x32_bf16(
                        false, fa.v, false, fb.v, (short)0, acc[j], false, false);
        }
    }

    #pragma unroll
    for (int j = 0; j < 4; ++j) {
        int col = tn * 64 + j * 16 + l16;
        float bv = bias ? bias[col] : 0.0f;
        #pragma unroll
        for (int r = 0; r < 8; ++r) {
            int row = tm * 16 + hi * 8 + r;      // C layout: M = hi*8 + r
            float v = acc[j][r] + bv;
            if (ACT == 1) v = 0.5f * v * (1.0f + erff(v * 0.70710678118654752f));
            size_t idx = (size_t)row * N + col;
            if (RES) v += resid[idx];
            if (Cf) Cf[idx] = v;
            if (Cb) Cb[idx] = f2bf(v);
        }
    }
}

// ---------------------------------------------------------------------------
// Init: agg = 0, mx = -inf, den = 0
// ---------------------------------------------------------------------------
__global__ void init_buffers_kernel(float* agg, float* mx, float* den,
                                    int nAgg, int nSeg) {
    int i = blockIdx.x * blockDim.x + threadIdx.x;
    if (i < nAgg) agg[i] = 0.0f;
    if (i < nSeg) { mx[i] = -__builtin_inff(); den[i] = 0.0f; }
}

__device__ __forceinline__ void atomicMaxF(float* a, float v) {
    if (v >= 0.0f) atomicMax((int*)a, __float_as_int(v));
    else           atomicMin((unsigned int*)a, __float_as_uint(v));
}

// ---------------------------------------------------------------------------
// Pass A: edge-bias MLP + per-head Q.K logits + segmented atomic max
// one thread per edge
// ---------------------------------------------------------------------------
__global__ void edge_logits_kernel(const long long* __restrict__ ei,
                                   const float* __restrict__ ef,
                                   const float* __restrict__ W1,
                                   const float* __restrict__ B1,
                                   const float* __restrict__ W2,
                                   const float* __restrict__ B2,
                                   const float* __restrict__ Q,
                                   const float* __restrict__ Kf,
                                   float* __restrict__ logits,
                                   float* __restrict__ mx,
                                   int E) {
    __shared__ float sW1[EDGE_F * EB_HID];
    __shared__ float sW2[EB_HID * N_HEADS];
    __shared__ float sB1[EB_HID];
    __shared__ float sB2[N_HEADS];
    for (int t = threadIdx.x; t < EDGE_F * EB_HID; t += blockDim.x) sW1[t] = W1[t];
    for (int t = threadIdx.x; t < EB_HID * N_HEADS; t += blockDim.x) sW2[t] = W2[t];
    if (threadIdx.x < EB_HID) sB1[threadIdx.x] = B1[threadIdx.x];
    if (threadIdx.x < N_HEADS) sB2[threadIdx.x] = B2[threadIdx.x];
    __syncthreads();

    int e = blockIdx.x * blockDim.x + threadIdx.x;
    if (e >= E) return;

    float f[EDGE_F];
    const float4* f4 = (const float4*)(ef + (size_t)e * EDGE_F);
    #pragma unroll
    for (int t = 0; t < 4; ++t) {
        float4 v = f4[t];
        f[4*t+0] = v.x; f[4*t+1] = v.y; f[4*t+2] = v.z; f[4*t+3] = v.w;
    }

    float bias[N_HEADS];
    #pragma unroll
    for (int o = 0; o < N_HEADS; ++o) bias[o] = sB2[o];
    for (int j = 0; j < EB_HID; ++j) {
        float hj = sB1[j];
        #pragma unroll
        for (int i = 0; i < EDGE_F; ++i) hj += f[i] * sW1[i * EB_HID + j];
        hj = hj > 0.0f ? hj : 0.1f * hj;                  // leaky relu
        #pragma unroll
        for (int o = 0; o < N_HEADS; ++o) bias[o] += hj * sW2[j * N_HEADS + o];
    }

    int src = (int)ei[2 * (size_t)e + 0];
    int dst = (int)ei[2 * (size_t)e + 1];
    const float4* q4 = (const float4*)(Q  + (size_t)dst * D_MODEL);
    const float4* k4 = (const float4*)(Kf + (size_t)src * D_MODEL);
    #pragma unroll
    for (int h = 0; h < N_HEADS; ++h) {
        float d = 0.0f;
        #pragma unroll
        for (int t = 0; t < 4; ++t) {
            float4 a = q4[h * 4 + t], b = k4[h * 4 + t];
            d += a.x * b.x + a.y * b.y + a.z * b.z + a.w * b.w;
        }
        float lg = d * 0.25f + bias[h];                   // 1/sqrt(DH)=0.25
        logits[(size_t)e * N_HEADS + h] = lg;
        atomicMaxF(&mx[(size_t)dst * N_HEADS + h], lg);
    }
}

// ---------------------------------------------------------------------------
// Pass B: ex = exp(logit - max); segmented atomic sum. thread per (edge,head)
// ---------------------------------------------------------------------------
__global__ void edge_exp_kernel(const long long* __restrict__ ei,
                                float* __restrict__ logits,
                                const float* __restrict__ mx,
                                float* __restrict__ den, int EH) {
    int i = blockIdx.x * blockDim.x + threadIdx.x;
    if (i >= EH) return;
    int e = i >> 3, h = i & 7;
    int dst = (int)ei[2 * (size_t)e + 1];
    size_t seg = (size_t)dst * N_HEADS + h;
    float ex = expf(logits[i] - mx[seg]);
    logits[i] = ex;
    atomicAdd(&den[seg], ex);
}

// ---------------------------------------------------------------------------
// Pass C: agg[dst] += alpha * V[src]. thread per (edge,head)
// ---------------------------------------------------------------------------
__global__ void edge_agg_kernel(const long long* __restrict__ ei,
                                const float* __restrict__ logits,
                                const float* __restrict__ den,
                                const float* __restrict__ V,
                                float* __restrict__ agg, int EH) {
    int i = blockIdx.x * blockDim.x + threadIdx.x;
    if (i >= EH) return;
    int e = i >> 3, h = i & 7;
    int src = (int)ei[2 * (size_t)e + 0];
    int dst = (int)ei[2 * (size_t)e + 1];
    float alpha = logits[i] / (den[(size_t)dst * N_HEADS + h] + 1e-12f);
    const float4* v4 = (const float4*)(V + (size_t)src * D_MODEL + h * DH);
    float* ag = agg + (size_t)dst * D_MODEL + h * DH;
    #pragma unroll
    for (int t = 0; t < 4; ++t) {
        float4 vv = v4[t];
        atomicAdd(&ag[4*t+0], alpha * vv.x);
        atomicAdd(&ag[4*t+1], alpha * vv.y);
        atomicAdd(&ag[4*t+2], alpha * vv.z);
        atomicAdd(&ag[4*t+3], alpha * vv.w);
    }
}

// ---------------------------------------------------------------------------
// f32 -> bf16 elementwise
// ---------------------------------------------------------------------------
__global__ void cvt_bf16_kernel(const float* __restrict__ a,
                                unsigned short* __restrict__ b, int n) {
    int i = blockIdx.x * blockDim.x + threadIdx.x;
    if (i < n) b[i] = f2bf(a[i]);
}

// ---------------------------------------------------------------------------
// Host-side launch
// ---------------------------------------------------------------------------
extern "C" void kernel_launch(void* const* d_in, const int* in_sizes, int n_in,
                              void* d_out, int out_size, void* d_ws, size_t ws_size,
                              hipStream_t stream) {
    const float*      x    = (const float*)d_in[0];
    const long long*  ei   = (const long long*)d_in[1];
    const float*      ef   = (const float*)d_in[2];
    const float*      Wq   = (const float*)d_in[3];
    const float*      Wk   = (const float*)d_in[4];
    const float*      Wv   = (const float*)d_in[5];
    const float*      Wo   = (const float*)d_in[6];
    const float*      ebW1 = (const float*)d_in[7];
    const float*      ebB1 = (const float*)d_in[8];
    const float*      ebW2 = (const float*)d_in[9];
    const float*      ebB2 = (const float*)d_in[10];
    const float*      ln1g = (const float*)d_in[11];
    const float*      ln1b = (const float*)d_in[12];
    const float*      ln2g = (const float*)d_in[13];
    const float*      ln2b = (const float*)d_in[14];
    const float*      fW1  = (const float*)d_in[15];
    const float*      fB1  = (const float*)d_in[16];
    const float*      fW2  = (const float*)d_in[17];
    const float*      fB2  = (const float*)d_in[18];
    float* out = (float*)d_out;

    const int Nn = N_NODES, E = N_EDGES;
    const size_t ND = (size_t)Nn * D_MODEL;       // 6.4M
    const size_t EH = (size_t)E * N_HEADS;        // 6.4M
    const size_t NH = (size_t)Nn * N_HEADS;       // 0.4M

    // workspace layout (256B aligned slabs)
    unsigned char* base = (unsigned char*)d_ws;
    size_t off = 0;
    auto alloc = [&](size_t bytes) -> void* {
        void* p = base + off;
        off = (off + bytes + 255) & ~(size_t)255;
        return p;
    };
    unsigned short* WqT  = (unsigned short*)alloc(128 * 128 * 2);
    unsigned short* WkT  = (unsigned short*)alloc(128 * 128 * 2);
    unsigned short* WvT  = (unsigned short*)alloc(128 * 128 * 2);
    unsigned short* WoT  = (unsigned short*)alloc(128 * 128 * 2);
    unsigned short* W1T  = (unsigned short*)alloc((size_t)FFN_HID * 128 * 2);
    unsigned short* W2T  = (unsigned short*)alloc((size_t)128 * FFN_HID * 2);
    unsigned short* yb   = (unsigned short*)alloc(ND * 2);
    float* Qb   = (float*)alloc(ND * 4);
    float* Kb   = (float*)alloc(ND * 4);
    float* Vb   = (float*)alloc(ND * 4);
    float* lgts = (float*)alloc(EH * 4);
    float* mx   = (float*)alloc(NH * 4);
    float* den  = (float*)alloc(NH * 4);
    float* agg  = (float*)alloc(ND * 4);
    unsigned short* aggb = (unsigned short*)alloc(ND * 2);
    float* x1   = (float*)alloc(ND * 4);
    unsigned short* zb   = (unsigned short*)alloc(ND * 2);
    unsigned short* h1b  = (unsigned short*)alloc((size_t)Nn * FFN_HID * 2);
    (void)ws_size;

    const int TB = 256;
    auto grid1 = [&](size_t n) { return (unsigned)((n + TB - 1) / TB); };

    // 1) weight prep: transpose + bf16
    transpose_to_bf16_kernel<<<grid1(128*128), TB, 0, stream>>>(Wq, WqT, 128, 128);
    transpose_to_bf16_kernel<<<grid1(128*128), TB, 0, stream>>>(Wk, WkT, 128, 128);
    transpose_to_bf16_kernel<<<grid1(128*128), TB, 0, stream>>>(Wv, WvT, 128, 128);
    transpose_to_bf16_kernel<<<grid1(128*128), TB, 0, stream>>>(Wo, WoT, 128, 128);
    transpose_to_bf16_kernel<<<grid1(128*FFN_HID), TB, 0, stream>>>(fW1, W1T, 128, FFN_HID);
    transpose_to_bf16_kernel<<<grid1((size_t)FFN_HID*128), TB, 0, stream>>>(fW2, W2T, FFN_HID, 128);

    // 2) LN1 -> bf16
    ln_bf16_kernel<<<grid1((size_t)Nn * 32), TB, 0, stream>>>(x, ln1g, ln1b, yb, Nn);

    // 3) Q,K,V projections (WMMA, 16x64 tiles)
    unsigned gemmBlocksD = (unsigned)(((Nn / 16) * (D_MODEL / 64) + 7) / 8);
    gemm_bf16_wmma_kernel<0,0><<<gemmBlocksD, TB, 0, stream>>>(yb, WqT, nullptr, nullptr, Qb, nullptr, Nn, D_MODEL, 128);
    gemm_bf16_wmma_kernel<0,0><<<gemmBlocksD, TB, 0, stream>>>(yb, WkT, nullptr, nullptr, Kb, nullptr, Nn, D_MODEL, 128);
    gemm_bf16_wmma_kernel<0,0><<<gemmBlocksD, TB, 0, stream>>>(yb, WvT, nullptr, nullptr, Vb, nullptr, Nn, D_MODEL, 128);

    // 4) init segmented-softmax + aggregation buffers
    init_buffers_kernel<<<grid1(ND), TB, 0, stream>>>(agg, mx, den, (int)ND, (int)NH);

    // 5) edge bias MLP + logits + segment max
    edge_logits_kernel<<<grid1(E), TB, 0, stream>>>(ei, ef, ebW1, ebB1, ebW2, ebB2,
                                                    Qb, Kb, lgts, mx, E);
    // 6) exp + segment sum
    edge_exp_kernel<<<grid1(EH), TB, 0, stream>>>(ei, lgts, mx, den, (int)EH);
    // 7) weighted scatter aggregation
    edge_agg_kernel<<<grid1(EH), TB, 0, stream>>>(ei, lgts, den, Vb, agg, (int)EH);

    // 8) agg -> bf16
    cvt_bf16_kernel<<<grid1(ND), TB, 0, stream>>>(agg, aggb, (int)ND);

    // 9) x1 = agg @ Wo + x   (WMMA, residual)
    gemm_bf16_wmma_kernel<0,1><<<gemmBlocksD, TB, 0, stream>>>(aggb, WoT, nullptr, x, x1, nullptr, Nn, D_MODEL, 128);

    // 10) LN2 -> bf16
    ln_bf16_kernel<<<grid1((size_t)Nn * 32), TB, 0, stream>>>(x1, ln2g, ln2b, zb, Nn);

    // 11) h1 = gelu(z @ W1 + b1) -> bf16 (WMMA)
    unsigned gemmBlocksH = (unsigned)(((Nn / 16) * (FFN_HID / 64) + 7) / 8);
    gemm_bf16_wmma_kernel<1,0><<<gemmBlocksH, TB, 0, stream>>>(zb, W1T, fB1, nullptr, nullptr, h1b, Nn, FFN_HID, 128);

    // 12) out = h1 @ W2 + b2 + x1 (WMMA, residual, f32 out)
    gemm_bf16_wmma_kernel<0,1><<<gemmBlocksD, TB, 0, stream>>>(h1b, W2T, fB2, x1, out, nullptr, Nn, D_MODEL, FFN_HID);
}
